// OTFeatureAligner_22917945491840
// MI455X (gfx1250) — matrix operand order repaired and checked by reading, not compiled, not run on hardware
//
#include <hip/hip_runtime.h>

typedef __attribute__((ext_vector_type(16))) _Float16 v16h;
typedef __attribute__((ext_vector_type(8)))  float    v8f;
typedef __attribute__((ext_vector_type(2)))  float    v2f;

#define B_  16
#define N_  4096
#define C_  512
#define M_  32

#if defined(__has_builtin)
#if __has_builtin(__builtin_amdgcn_s_wait_asynccnt)
#define WAIT_ASYNC(n) __builtin_amdgcn_s_wait_asynccnt(n)
#endif
#endif
#ifndef WAIT_ASYNC
#define WAIT_ASYNC(n) asm volatile("s_wait_asynccnt %0" :: "i"(n) : "memory")
#endif

// async copy of 16 bytes global -> LDS (per active lane), tracked by ASYNCcnt
__device__ __forceinline__ void async_b128(const void* gptr, void* lptr) {
  unsigned lds = (unsigned)(uintptr_t)lptr;
  asm volatile("global_load_async_to_lds_b128 %0, %1, off"
               :: "v"(lds), "v"(gptr) : "memory");
}

__device__ inline float waveReduceSum(float v) {
  #pragma unroll
  for (int m = 16; m > 0; m >>= 1) v += __shfl_xor(v, m, 32);
  return v;
}

// ---------------- init: u=0, v=0, errPart=0, done=0 ----------------
__global__ void ot_init(float* u, float* vvec, float* errPart, int* done) {
  int idx = blockIdx.x * 256 + threadIdx.x;
  if (idx < B_ * N_) u[idx] = 0.0f;
  if (idx < B_ * M_) vvec[idx] = 0.0f;
  if (idx < B_)      errPart[idx] = 0.0f;
  if (idx == 0)      *done = 0;
}

// ---------------- nu from argmax of score_map ----------------
__global__ void ot_nu(const float* __restrict__ score, float* __restrict__ nu) {
  int b = blockIdx.x, tid = threadIdx.x;
  __shared__ int cnt[M_];
  if (tid < M_) cnt[tid] = 0;
  __syncthreads();
  for (int i = 0; i < 16; ++i) {
    int p = i * 256 + tid;
    const float* sp = score + (size_t)b * M_ * N_ + p;
    float best = sp[0];
    int arg = 0;
    #pragma unroll
    for (int k = 1; k < M_; ++k) {
      float v = sp[(size_t)k * N_];
      if (v > best) { best = v; arg = k; }
    }
    atomicAdd(&cnt[arg], 1);
  }
  __syncthreads();
  if (tid < M_) {
    float raw = (float)cnt[tid] / (4096.0f + 1e-8f) + 1e-6f;
    float tot = waveReduceSum(raw);
    nu[b * M_ + tid] = raw / tot;
  }
}

// ---------------- inverse row norms (X rows then Y rows) ----------------
__global__ void ot_norm(const float* __restrict__ X, const float* __restrict__ Y,
                        float* __restrict__ invx, float* __restrict__ invy) {
  int g = blockIdx.x * 8 + (threadIdx.x >> 5);
  int lane = threadIdx.x & 31;
  const int RX = B_ * N_;
  const int RT = RX + B_ * M_;
  if (g >= RT) return;
  const float* p = (g < RX) ? (X + (size_t)g * C_) : (Y + (size_t)(g - RX) * C_);
  float s = 0.0f;
  #pragma unroll
  for (int i = 0; i < 16; ++i) { float v = p[lane + i * 32]; s += v * v; }
  s = waveReduceSum(s);
  if (lane == 0) {
    float r = rsqrtf(s);
    if (g < RX) invx[g] = r; else invy[g - RX] = r;
  }
}

// ---------------- W_top -> transposed f16 [n][k] ----------------
__global__ void ot_wconv(const float* __restrict__ W, _Float16* __restrict__ WtT) {
  int idx = blockIdx.x * 256 + threadIdx.x;      // idx = k*512 + n, k<512
  int k = idx >> 9, n = idx & 511;
  WtT[(size_t)n * C_ + k] = (_Float16)W[idx];
}

// ---------------- YW^T f16: YWT[b][co][m] = sum_c Y[b,m,c]*W[512+c][co] ----------------
__global__ void ot_yw(const float* __restrict__ Y, const float* __restrict__ W,
                      _Float16* __restrict__ YWT) {
  int bm = blockIdx.x;                 // 0..511
  int b = bm >> 5, m = bm & 31;
  int co = blockIdx.y * 256 + threadIdx.x;
  const float* yr = Y + (size_t)bm * C_;
  float acc = 0.0f;
  for (int c = 0; c < C_; ++c)
    acc += yr[c] * W[(size_t)(C_ + c) * C_ + co];
  YWT[((size_t)(b * C_ + co)) * M_ + m] = (_Float16)acc;
}

// ---------------- cost + Kexp via V_WMMA_F32_16X16X4_F32, async-LDS double buffer ----
__global__ void ot_cost(const float* __restrict__ X, const float* __restrict__ Y,
                        const float* __restrict__ invx, const float* __restrict__ invy,
                        float* __restrict__ cost, float* __restrict__ Kexp) {
  int b = blockIdx.x, n0 = blockIdx.y * 64;
  int tid = threadIdx.x, lane = tid & 31, wave = tid >> 5;
  int wrow = wave & 3, m0 = (wave >> 2) * 16;
  __shared__ float Xs[2][64 * 32];
  __shared__ float Ys[2][32 * 32];
  v8f acc = {};
  int koff = (lane < 16) ? 0 : 2;
  int arow = wrow * 16 + (lane & 15);
  int bcol = m0 + (lane & 15);

  const float* Xb = X + (size_t)(b * N_ + n0) * C_;
  const float* Yb = Y + (size_t)(b * M_) * C_;
  int rx = tid >> 3;            // 0..31 (row block; +32 for second copy)
  int c4 = (tid & 7) * 4;       // float column within 32-wide tile

  // stage one k-slice (32 wide) into buffer `buf`: 3 async b128 per wave
  auto stage = [&](int buf, int kk) {
    async_b128(Xb + (size_t)rx * C_ + kk + c4,        &Xs[buf][4 * tid]);
    async_b128(Xb + (size_t)(rx + 32) * C_ + kk + c4, &Xs[buf][4 * (tid + 256)]);
    async_b128(Yb + (size_t)rx * C_ + kk + c4,        &Ys[buf][4 * tid]);
  };

  stage(0, 0);
  for (int it = 0; it < 16; ++it) {
    int buf = it & 1;
    if (it < 15) {
      stage(buf ^ 1, (it + 1) * 32);
      WAIT_ASYNC(3);            // current buffer's 3 copies done; next 3 in flight
    } else {
      WAIT_ASYNC(0);
    }
    __syncthreads();
    #pragma unroll
    for (int kin = 0; kin < 32; kin += 4) {
      v2f a, bb;
      a.x  = Xs[buf][arow * 32 + kin + koff];
      a.y  = Xs[buf][arow * 32 + kin + koff + 1];
      bb.x = Ys[buf][bcol * 32 + kin + koff];
      bb.y = Ys[buf][bcol * 32 + kin + koff + 1];
      acc = __builtin_amdgcn_wmma_f32_16x16x4_f32(false, a, false, bb,
                                                  (short)0, acc, false, false);
    }
    __syncthreads();
  }
  #pragma unroll
  for (int j = 0; j < 8; ++j) {
    int ng = n0 + wrow * 16 + j + ((lane < 16) ? 0 : 8);
    int mg = bcol;
    float cv = 1.0f - acc[j] * invx[b * N_ + ng] * invy[b * M_ + mg];
    size_t idx = ((size_t)(b * N_ + ng)) * M_ + mg;
    cost[idx] = cv;
    Kexp[idx] = expf(-cv * 20.0f);   // exp(-cost/0.05)
  }
}

// ---------------- Sinkhorn u-update ----------------
__global__ void ot_uk(const float* __restrict__ Kexp, float* __restrict__ u,
                      const float* __restrict__ vvec, float* __restrict__ errPart,
                      const int* __restrict__ done) {
  if (*done) return;
  int b = blockIdx.x, tid = threadIdx.x;
  __shared__ float vs[M_];
  __shared__ float red[256];
  if (tid < M_) vs[tid] = vvec[b * M_ + tid];
  __syncthreads();
  float errL = 0.0f;
  for (int i = 0; i < 16; ++i) {
    int n = i * 256 + tid;
    const float* kr = Kexp + ((size_t)(b * N_ + n)) * M_;
    float s = 0.0f;
    #pragma unroll
    for (int m = 0; m < M_; ++m) s += kr[m] * vs[m];
    float un = (1.0f / N_) / (s + 1e-8f);
    float uo = u[b * N_ + n];
    errL += fabsf(un - uo);
    u[b * N_ + n] = un;
  }
  red[tid] = errL;
  __syncthreads();
  for (int s = 128; s > 0; s >>= 1) {
    if (tid < s) red[tid] += red[tid + s];
    __syncthreads();
  }
  if (tid == 0) errPart[b] = red[0];
}

// ---------------- Sinkhorn v-update + done latch ----------------
__global__ void ot_vk(const float* __restrict__ Kexp, const float* __restrict__ u,
                      float* __restrict__ vvec, const float* __restrict__ nu,
                      const float* __restrict__ errPart, int* __restrict__ done) {
  if (*done) return;
  int b = blockIdx.x, tid = threadIdx.x;
  __shared__ float part[8][M_];
  int m = tid & 31, ch = tid >> 5;
  float s = 0.0f;
  int n0 = ch * 512;
  for (int n = n0; n < n0 + 512; ++n)
    s += Kexp[((size_t)(b * N_ + n)) * M_ + m] * u[b * N_ + n];
  part[ch][m] = s;
  __syncthreads();
  if (tid < M_) {
    float t = 0.0f;
    #pragma unroll
    for (int c = 0; c < 8; ++c) t += part[c][tid];
    vvec[b * M_ + tid] = nu[b * M_ + tid] / (t + 1e-8f);
  }
  __syncthreads();
  if (b == 0 && tid == 0) {
    float e = 0.0f;
    #pragma unroll
    for (int i = 0; i < B_; ++i) e += errPart[i];
    if (e * (1.0f / B_) < 0.1f) *done = 1;
  }
}

// ---------------- pi, invrow, OT-loss partials ----------------
__global__ void ot_pi(const float* __restrict__ Kexp, const float* __restrict__ cost,
                      const float* __restrict__ u, const float* __restrict__ vvec,
                      float* __restrict__ pi, float* __restrict__ invrow,
                      float* __restrict__ lossPart) {
  int b = blockIdx.x, ch = blockIdx.y, tid = threadIdx.x;
  __shared__ float vs[M_];
  __shared__ float red[256];
  if (tid < M_) vs[tid] = vvec[b * M_ + tid];
  __syncthreads();
  int n = ch * 256 + tid;
  size_t base = ((size_t)(b * N_ + n)) * M_;
  float uu = u[b * N_ + n];
  float rs = 0.0f, ot = 0.0f;
  #pragma unroll
  for (int m = 0; m < M_; ++m) {
    float p = uu * Kexp[base + m] * vs[m];
    pi[base + m] = p;
    rs += p;
    ot += p * cost[base + m];
  }
  invrow[b * N_ + n] = 1.0f / (rs + 1e-8f);
  red[tid] = ot;
  __syncthreads();
  for (int s = 128; s > 0; s >>= 1) {
    if (tid < s) red[tid] += red[tid + s];
    __syncthreads();
  }
  if (tid == 0) lossPart[b * 16 + ch] = red[0];
}

__global__ void ot_lossfin(const float* __restrict__ lossPart, float* __restrict__ loss) {
  int t = threadIdx.x;
  if (t < B_) {
    float s = 0.0f;
    #pragma unroll
    for (int i = 0; i < 16; ++i) s += lossPart[t * 16 + i];
    loss[t] = s;
  }
}

// ---------------- fused X@Wtop + pi_norm@YW + bias + LayerNorm ----------------
__global__ void ot_fuse(const float* __restrict__ X, const float* __restrict__ pi,
                        const float* __restrict__ invrow,
                        const _Float16* __restrict__ WtT, const _Float16* __restrict__ YWT,
                        const float* __restrict__ bias, const float* __restrict__ gamma,
                        const float* __restrict__ beta, float* __restrict__ out) {
  int b = blockIdx.x, r0 = blockIdx.y * 16;
  int tid = threadIdx.x, lane = tid & 31, wave = tid >> 5;
  __shared__ float   Xsf[2][16 * 32];   // f32 k-slices, async staged
  __shared__ _Float16 Ap[16 * 32];
  __shared__ float   Co[16 * 520];

  // stage pi_norm tile (16 rows x 32 m) as f16
  for (int e = tid; e < 512; e += 256) {
    int r = e >> 5, m = e & 31;
    Ap[e] = (_Float16)(pi[((size_t)(b * N_ + r0 + r)) * M_ + m] * invrow[b * N_ + r0 + r]);
  }

  v8f acc[4] = {};
  int row   = lane & 15;
  int hbase = (lane < 16) ? 0 : 16;   // A-fragment K offset
  int kb16  = (lane < 16) ? 0 : 16;   // B-fragment K offset
  int c0    = wave * 64;

  const float* Xb = X + (size_t)(b * N_ + r0) * C_;
  int rx = tid >> 3;            // 0..31 (only tid<128 used)
  int c4 = (tid & 7) * 4;
  bool stager = (tid < 128);    // waves 0-3 do the staging (1 async b128 each)

  auto stage = [&](int buf, int kk) {
    if (stager)
      async_b128(Xb + (size_t)rx * C_ + kk + c4, &Xsf[buf][4 * tid]);
  };

  stage(0, 0);
  for (int it = 0; it < 16; ++it) {
    int buf = it & 1;
    int kk = it * 32;
    if (it < 15) {
      stage(buf ^ 1, kk + 32);
      WAIT_ASYNC(1);
    } else {
      WAIT_ASYNC(0);
    }
    __syncthreads();
    v16h af;
    #pragma unroll
    for (int h = 0; h < 16; ++h) {
      int k = h + hbase;
      af[h] = (_Float16)Xsf[buf][row * 32 + k];
    }
    #pragma unroll
    for (int t = 0; t < 4; ++t) {
      const _Float16* wp = WtT + ((size_t)(c0 + t * 16 + row)) * C_ + kk + kb16;
      v16h bf = *(const v16h*)wp;
      acc[t] = __builtin_amdgcn_wmma_f32_16x16x32_f16(false, af, false, bf,
                                                      (short)0, acc[t], false, false);
    }
    __syncthreads();
  }

  // pi_norm @ YW contribution (K = 32 exactly one WMMA per tile)
  v16h apf;
  #pragma unroll
  for (int h = 0; h < 16; ++h) {
    int k = h + hbase;
    apf[h] = Ap[row * 32 + k];
  }
  #pragma unroll
  for (int t = 0; t < 4; ++t) {
    const _Float16* yp = YWT + ((size_t)(b * C_ + c0 + t * 16 + row)) * M_ + kb16;
    v16h bf = *(const v16h*)yp;
    acc[t] = __builtin_amdgcn_wmma_f32_16x16x32_f16(false, apf, false, bf,
                                                    (short)0, acc[t], false, false);
  }

  // write tiles + bias to LDS
  int rofs = (lane < 16) ? 0 : 8;
  #pragma unroll
  for (int t = 0; t < 4; ++t) {
    int col = c0 + t * 16 + row;
    #pragma unroll
    for (int j = 0; j < 8; ++j)
      Co[(j + rofs) * 520 + col] = acc[t][j] + bias[col];
  }
  __syncthreads();

  // LayerNorm: each wave handles 2 rows of the 16-row block
  for (int rr = 0; rr < 2; ++rr) {
    int r = wave * 2 + rr;
    float s = 0.0f, sq = 0.0f;
    #pragma unroll
    for (int i = 0; i < 16; ++i) {
      float vv = Co[r * 520 + lane + i * 32];
      s += vv; sq += vv * vv;
    }
    s = waveReduceSum(s);
    sq = waveReduceSum(sq);
    float mean = s * (1.0f / 512.0f);
    float var  = sq * (1.0f / 512.0f) - mean * mean;
    float inv  = rsqrtf(var + 1e-5f);
    #pragma unroll
    for (int i = 0; i < 16; ++i) {
      int col = lane + i * 32;
      float vv = Co[r * 520 + col];
      out[((size_t)(b * N_ + r0 + r)) * C_ + col] = (vv - mean) * inv * gamma[col] + beta[col];
    }
  }
}

extern "C" void kernel_launch(void* const* d_in, const int* in_sizes, int n_in,
                              void* d_out, int out_size, void* d_ws, size_t ws_size,
                              hipStream_t stream) {
  (void)in_sizes; (void)n_in; (void)out_size; (void)ws_size;
  const float* X     = (const float*)d_in[0];   // [B,N,C]
  const float* Y     = (const float*)d_in[1];   // [B,M,C]
  const float* score = (const float*)d_in[2];   // [B,M,64,64]
  const float* W     = (const float*)d_in[3];   // [2C,C]
  const float* bias  = (const float*)d_in[4];
  const float* gamma = (const float*)d_in[5];
  const float* beta  = (const float*)d_in[6];

  float* out  = (float*)d_out;
  float* loss = out + (size_t)B_ * N_ * C_;
  float* pi   = loss + B_;

  char* w = (char*)d_ws;
  auto alloc = [&](size_t bytes) -> char* {
    char* p = w;
    w += (bytes + 255) & ~(size_t)255;
    return p;
  };
  float* cost    = (float*)alloc(sizeof(float) * B_ * N_ * M_);
  float* Kexp    = (float*)alloc(sizeof(float) * B_ * N_ * M_);
  float* u       = (float*)alloc(sizeof(float) * B_ * N_);
  float* invrow  = (float*)alloc(sizeof(float) * B_ * N_);
  float* invx    = (float*)alloc(sizeof(float) * B_ * N_);
  float* invy    = (float*)alloc(sizeof(float) * B_ * M_);
  float* vvec    = (float*)alloc(sizeof(float) * B_ * M_);
  float* nu      = (float*)alloc(sizeof(float) * B_ * M_);
  float* errPart = (float*)alloc(sizeof(float) * B_);
  float* lossPart= (float*)alloc(sizeof(float) * B_ * 16);
  int*   done    = (int*)alloc(sizeof(int));
  _Float16* WtT  = (_Float16*)alloc(sizeof(_Float16) * C_ * C_);
  _Float16* YWT  = (_Float16*)alloc(sizeof(_Float16) * B_ * C_ * M_);

  ot_init<<<256, 256, 0, stream>>>(u, vvec, errPart, done);
  ot_nu<<<B_, 256, 0, stream>>>(score, nu);
  ot_norm<<<(B_ * N_ + B_ * M_) / 8, 256, 0, stream>>>(X, Y, invx, invy);
  ot_wconv<<<1024, 256, 0, stream>>>(W, WtT);
  ot_yw<<<dim3(B_ * M_, 2), 256, 0, stream>>>(Y, W, YWT);
  ot_cost<<<dim3(B_, N_ / 64), 256, 0, stream>>>(X, Y, invx, invy, cost, Kexp);
  for (int it = 0; it < 50; ++it) {
    ot_uk<<<B_, 256, 0, stream>>>(Kexp, u, vvec, errPart, done);
    ot_vk<<<B_, 256, 0, stream>>>(Kexp, u, vvec, nu, errPart, done);
  }
  ot_pi<<<dim3(B_, 16), 256, 0, stream>>>(Kexp, cost, u, vvec, pi, invrow, lossPart);
  ot_lossfin<<<1, 256, 0, stream>>>(lossPart, loss);
  ot_fuse<<<dim3(B_, N_ / 16), 256, 0, stream>>>(X, pi, invrow, WtT, YWT,
                                                 bias, gamma, beta, out);
}
